// GAT_52656299049251
// MI455X (gfx1250) — matrix operand order repaired and checked by reading, not compiled, not run on hardware
//
#include <hip/hip_runtime.h>
#include <hip/hip_bf16.h>

// ---------------------------------------------------------------------------
// GAT forward (3x GATConv + BN + ELU + residual, tri-pool, 2 linear heads)
// GEMMs run on v_wmma_f32_16x16x32_bf16 (CDNA5 WMMA, wave32), double-buffered.
// Edge softmax/aggregation runs on fp32 VALU + global atomics (L2-resident).
// ---------------------------------------------------------------------------

#define WG 256

static constexpr int NN   = 50000;
static constexpr int EE   = 800000;
static constexpr int GG   = 512;
static constexpr int HID  = 256;
static constexpr int HEADS = 8;
static constexpr int CH   = 32;
static constexpr unsigned ENC_NEG_INF = 0x007FFFFFu; // fenc(-inf)

typedef __attribute__((ext_vector_type(16))) __bf16    v16bf;
typedef __attribute__((ext_vector_type(8)))  float     v8f;
typedef __attribute__((ext_vector_type(8)))  unsigned  v8u;

union ABu { v8u u; v16bf v; };

// ---- helpers ---------------------------------------------------------------

__device__ __forceinline__ unsigned fenc(float x) {
    unsigned u = __float_as_uint(x);
    return (u & 0x80000000u) ? ~u : (u | 0x80000000u);
}
__device__ __forceinline__ float fdec(unsigned e) {
    unsigned u = (e & 0x80000000u) ? (e & 0x7FFFFFFFu) : ~e;
    return __uint_as_float(u);
}
__device__ __forceinline__ unsigned bf16pair(float a, float b) {
    unsigned ua = __float_as_uint(a), ub = __float_as_uint(b);
    ua += 0x7FFFu + ((ua >> 16) & 1u);   // RNE to bf16
    ub += 0x7FFFu + ((ub >> 16) & 1u);
    return (ua >> 16) | (ub & 0xFFFF0000u);
}
__device__ __forceinline__ void edge_sd(const int* __restrict__ ei, int e,
                                        int& s, int& d) {
    if (e < EE) { s = ei[e]; d = ei[EE + e]; }
    else        { s = d = e - EE; }          // self-loop tail
}

// ---- generic fill ----------------------------------------------------------

__global__ void k_fill_u32(unsigned* __restrict__ p, unsigned v, int n) {
    int i = blockIdx.x * WG + threadIdx.x;
    if (i < n) p[i] = v;
}

// ---- WMMA fragment packing -------------------------------------------------
// A (16x32 bf16): lanes 0-15 row M=lane, K in {0..15};
// lanes 16-31 row M=lane-16, K in {16..31}; dword v packs 2 K values.
__global__ void k_pack_A(const float* __restrict__ F, unsigned* __restrict__ Apk,
                         int N, int D) {
    int tid = blockIdx.x * WG + threadIdx.x;
    int KT = D >> 5;
    int total = ((N + 15) >> 4) * KT * 256;
    if (tid >= total) return;
    int v    = tid & 7;
    int lane = (tid >> 3) & 31;
    int t    = tid >> 8;
    int kt   = t % KT;
    int rt   = t / KT;
    int row  = rt * 16 + (lane & 15);
    int kb   = (lane < 16) ? 0 : 16;
    int koff = kb + 2 * v;
    int K    = kt * 32 + koff;
    float f0 = (row < N) ? F[(size_t)row * D + K]     : 0.0f;
    float f1 = (row < N) ? F[(size_t)row * D + K + 1] : 0.0f;
    Apk[tid] = bf16pair(f0, f1);
}

// B (32x16 bf16): lanes 0-15 K=0..15, lanes 16-31 K=16..31, col = lane&15,
// dword v packs K = kgrp + 2v, +1.
__global__ void k_pack_W(const float* __restrict__ W, unsigned* __restrict__ Wpk,
                         int D) {
    int tid = blockIdx.x * WG + threadIdx.x;
    int KT = D >> 5;
    int total = KT * 16 * 256;
    if (tid >= total) return;
    int v    = tid & 7;
    int lane = (tid >> 3) & 31;
    int t    = tid >> 8;
    int ct   = t % 16;
    int kt   = t / 16;
    int col  = ct * 16 + (lane & 15);
    int kg   = (lane < 16) ? 0 : 16;
    int K    = kt * 32 + kg + 2 * v;
    float f0 = W[(size_t)K * HID + col];
    float f1 = W[(size_t)(K + 1) * HID + col];
    Wpk[tid] = bf16pair(f0, f1);
}

// ---- GEMM: C[N,256] = A[N,D] * W[D,256] via WMMA bf16 ----------------------
// Block = 32-row strip. 8 waves in a 2x4 grid: waves 0-3 rows 0-15,
// waves 4-7 rows 16-31; each wave owns 4 of the 16 column tiles.
// Double-buffered fragments so loads for kt+1 are in flight across the
// WMMAs for kt (avoids s_wait_loadcnt 0 in front of every WMMA).
__global__ __launch_bounds__(256) void k_gemm(const unsigned* __restrict__ Apk,
                                              const unsigned* __restrict__ Wpk,
                                              float* __restrict__ C,
                                              int N, int KT) {
    int wave = threadIdx.x >> 5;
    int lane = threadIdx.x & 31;
    int RT   = (N + 15) >> 4;
    int rt   = blockIdx.x * 2 + (wave >> 2);
    if (rt >= RT) rt = RT - 1;              // tail: redundant compute, guarded store
    int cg   = (wave & 3) * 4;              // first of this wave's 4 col tiles

    const unsigned* abase = Apk + (size_t)rt * KT * 256 + lane * 8;
    const unsigned* bbase = Wpk + (size_t)cg * 256 + lane * 8;
    // A frag (kt):      abase + kt*256
    // B frag (kt, j):   bbase + kt*4096 + j*256

    v8f acc[4] = {{}, {}, {}, {}};
    ABu aA, aB, bA[4], bB[4];

    aA.u = *(const v8u*)(abase);
#pragma unroll
    for (int j = 0; j < 4; ++j) bA[j].u = *(const v8u*)(bbase + (size_t)j * 256);

    for (int kt = 0; kt < KT; kt += 2) {
        // stage kt+1 (KT is even, always valid)
        aB.u = *(const v8u*)(abase + (size_t)(kt + 1) * 256);
#pragma unroll
        for (int j = 0; j < 4; ++j)
            bB[j].u = *(const v8u*)(bbase + (size_t)(kt + 1) * 4096 + j * 256);
        __builtin_prefetch(abase + (size_t)(kt + 4) * 256, 0, 0);
#pragma unroll
        for (int j = 0; j < 4; ++j)
            acc[j] = __builtin_amdgcn_wmma_f32_16x16x32_bf16(
                         false, aA.v, false, bA[j].v, (short)0, acc[j], false, false);
        if (kt + 2 < KT) {                   // stage kt+2
            aA.u = *(const v8u*)(abase + (size_t)(kt + 2) * 256);
#pragma unroll
            for (int j = 0; j < 4; ++j)
                bA[j].u = *(const v8u*)(bbase + (size_t)(kt + 2) * 4096 + j * 256);
        }
#pragma unroll
        for (int j = 0; j < 4; ++j)
            acc[j] = __builtin_amdgcn_wmma_f32_16x16x32_bf16(
                         false, aB.v, false, bB[j].v, (short)0, acc[j], false, false);
    }

    // C/D layout: VGPR r -> M = r (lanes 0-15) / 8+r (lanes 16-31), N = lane&15
    int rbase = rt * 16 + ((lane >> 4) << 3);
#pragma unroll
    for (int j = 0; j < 4; ++j) {
        int col = (cg + j) * 16 + (lane & 15);
        for (int r = 0; r < 8; ++r) {
            int row = rbase + r;
            if (row < N) C[(size_t)row * HID + col] = acc[j][r];
        }
    }
}

// ---- attention scores: s = sum_c h[n,h,c] * a[h,c] -------------------------
__global__ void k_scores(const float* __restrict__ hP,
                         const float* __restrict__ asrc,
                         const float* __restrict__ adst,
                         float* __restrict__ ss, float* __restrict__ sd, int N) {
    int tid = blockIdx.x * WG + threadIdx.x;
    if (tid >= N * HEADS) return;
    int n = tid >> 3, h = tid & 7;
    const float* hp = hP + (size_t)n * HID + h * CH;
    float a = 0.f, b = 0.f;
    for (int c = 0; c < CH; ++c) {
        float x = hp[c];
        a += x * asrc[h * CH + c];
        b += x * adst[h * CH + c];
    }
    ss[tid] = a; sd[tid] = b;
}

// ---- segment softmax over incoming edges -----------------------------------
__global__ void k_edge_max(const int* __restrict__ ei,
                           const float* __restrict__ ss,
                           const float* __restrict__ sd,
                           unsigned* __restrict__ mxk) {
    int tid = blockIdx.x * WG + threadIdx.x;
    if (tid >= (EE + NN) * HEADS) return;
    int e = tid >> 3, h = tid & 7, s, d;
    edge_sd(ei, e, s, d);
    float al = ss[s * HEADS + h] + sd[d * HEADS + h];
    al = al > 0.f ? al : 0.2f * al;
    atomicMax(&mxk[d * HEADS + h], fenc(al));
}

__global__ void k_edge_exp(const int* __restrict__ ei,
                           const float* __restrict__ ss,
                           const float* __restrict__ sd,
                           const unsigned* __restrict__ mxk,
                           float* __restrict__ ebuf, float* __restrict__ Z) {
    int tid = blockIdx.x * WG + threadIdx.x;
    if (tid >= (EE + NN) * HEADS) return;
    int e = tid >> 3, h = tid & 7, s, d;
    edge_sd(ei, e, s, d);
    float al = ss[s * HEADS + h] + sd[d * HEADS + h];
    al = al > 0.f ? al : 0.2f * al;
    float ev = __expf(al - fdec(mxk[d * HEADS + h]));
    ebuf[tid] = ev;
    atomicAdd(&Z[d * HEADS + h], ev);
}

// one thread = (edge, 4 consecutive channels in one head): b128 gather.
__global__ void k_edge_agg(const int* __restrict__ ei,
                           const float* __restrict__ ebuf,
                           const float* __restrict__ Z,
                           const float* __restrict__ hP,
                           float* __restrict__ hO) {
    int tid = blockIdx.x * WG + threadIdx.x;
    if (tid >= (EE + NN) * (HID / 4)) return;
    int e = tid >> 6, ch4 = (tid & 63) * 4, s, d;
    edge_sd(ei, e, s, d);
    int h = ch4 >> 5;
    float coef = ebuf[(size_t)e * HEADS + h] / (Z[d * HEADS + h] + 1e-16f);
    const float4 msg = *(const float4*)(hP + (size_t)s * HID + ch4);
    float* o = hO + (size_t)d * HID + ch4;
    atomicAdd(o + 0, msg.x * coef);
    atomicAdd(o + 1, msg.y * coef);
    atomicAdd(o + 2, msg.z * coef);
    atomicAdd(o + 3, msg.w * coef);
}

// ---- bias + batchnorm + residual + ELU -------------------------------------
__global__ void k_post(const float* __restrict__ hO, const float* __restrict__ b,
                       const float* __restrict__ g, const float* __restrict__ be,
                       const float* __restrict__ m, const float* __restrict__ v,
                       const float* __restrict__ hin, float* __restrict__ hout,
                       int N, int residual) {
    int tid = blockIdx.x * WG + threadIdx.x;
    if (tid >= N * HID) return;
    int ch = tid & 255;
    float val = hO[tid] + b[ch];
    val = (val - m[ch]) * rsqrtf(v[ch] + 1e-5f) * g[ch] + be[ch];
    if (residual) val += hin[tid];
    hout[tid] = val > 0.f ? val : (__expf(val) - 1.f);
}

// ---- pooling ---------------------------------------------------------------
__global__ void k_pool_cnt(const int* __restrict__ batch, float* __restrict__ cnt,
                           int N) {
    int i = blockIdx.x * WG + threadIdx.x;
    if (i < N) atomicAdd(&cnt[batch[i]], 1.0f);
}

__global__ void k_pool_acc(const float* __restrict__ hA,
                           const int* __restrict__ batch,
                           float* __restrict__ sums, unsigned* __restrict__ mxp,
                           int N) {
    int tid = blockIdx.x * WG + threadIdx.x;
    if (tid >= N * HID) return;
    int n = tid >> 8, ch = tid & 255;
    int gb = batch[n];
    float val = hA[tid];
    atomicAdd(&sums[gb * HID + ch], val);
    atomicMax(&mxp[gb * HID + ch], fenc(val));
}

__global__ void k_pooled(const float* __restrict__ sums,
                         const float* __restrict__ cnt,
                         const unsigned* __restrict__ mxp,
                         float* __restrict__ pooled) {
    int tid = blockIdx.x * WG + threadIdx.x;
    if (tid >= GG * HID) return;
    int gb = tid >> 8, ch = tid & 255;
    float c = cnt[gb];
    float s = sums[tid];
    float mean = s / fmaxf(c, 1.0f);
    float mv = fdec(mxp[tid]);
    if (!(mv > -3.0e38f)) mv = 0.0f;   // non-finite -> 0
    float* p = pooled + (size_t)gb * (3 * HID);
    p[ch]           = mean;
    p[HID + ch]     = s;
    p[2 * HID + ch] = mv;
}

__global__ void k_heads(const float* __restrict__ pooled,
                        const float* __restrict__ ow0, const float* __restrict__ ob0,
                        const float* __restrict__ ow1, const float* __restrict__ ob1,
                        float* __restrict__ out) {
    int tid = blockIdx.x * WG + threadIdx.x;
    if (tid >= GG * 6) return;
    int gb = tid / 6, k = tid % 6, i = k / 3, c = k % 3;
    const float* ow = i ? ow1 : ow0;
    const float* ob = i ? ob1 : ob0;
    const float* p = pooled + (size_t)gb * (3 * HID);
    float acc = ob[c];
    for (int j = 0; j < 3 * HID; ++j) acc += p[j] * ow[j * 3 + c];
    out[i * (GG * 3) + gb * 3 + c] = acc;
}

// ---------------------------------------------------------------------------

static inline dim3 gsz(long long n) { return dim3((unsigned)((n + WG - 1) / WG)); }

extern "C" void kernel_launch(void* const* d_in, const int* in_sizes, int n_in,
                              void* d_out, int out_size, void* d_ws, size_t ws_size,
                              hipStream_t stream) {
    const float* x     = (const float*)d_in[0];
    const int*   ei    = (const int*)d_in[1];
    const int*   batch = (const int*)d_in[2];
    const float* ow0 = (const float*)d_in[27];
    const float* ob0 = (const float*)d_in[28];
    const float* ow1 = (const float*)d_in[29];
    const float* ob1 = (const float*)d_in[30];

    struct Lp { const float *W, *as_, *ad, *b, *g, *be, *m, *v; };
    Lp L[3];
    for (int l = 0; l < 3; ++l) {
        int base = 3 + l * 8;
        L[l].W   = (const float*)d_in[base + 0];
        L[l].as_ = (const float*)d_in[base + 1];
        L[l].ad  = (const float*)d_in[base + 2];
        L[l].b   = (const float*)d_in[base + 3];
        L[l].g   = (const float*)d_in[base + 4];
        L[l].be  = (const float*)d_in[base + 5];
        L[l].m   = (const float*)d_in[base + 6];
        L[l].v   = (const float*)d_in[base + 7];
    }

    // ---- workspace carve (256B aligned) ----
    size_t off = 0;
    auto carve = [&](size_t bytes) -> void* {
        void* p = (char*)d_ws + off;
        off += (bytes + 255) & ~(size_t)255;
        return p;
    };
    const int RT = (NN + 15) / 16;
    unsigned* Wpk1 = (unsigned*)carve((size_t)4 * 16 * 256 * 4);
    unsigned* Wpk2 = (unsigned*)carve((size_t)8 * 16 * 256 * 4);
    unsigned* Wpk3 = (unsigned*)carve((size_t)8 * 16 * 256 * 4);
    // Apk and ebuf have disjoint live ranges inside a layer -> share storage
    size_t apk_b  = (size_t)RT * 8 * 256 * 4;
    size_t ebuf_b = (size_t)(EE + NN) * HEADS * 4;
    void*  shared = carve(apk_b > ebuf_b ? apk_b : ebuf_b);
    unsigned* Apk  = (unsigned*)shared;
    float*    ebuf = (float*)shared;
    float* hP  = (float*)carve((size_t)NN * HID * 4);
    float* hA  = (float*)carve((size_t)NN * HID * 4);
    float* hO  = (float*)carve((size_t)NN * HID * 4);
    float* ss  = (float*)carve((size_t)NN * HEADS * 4);
    float* sd  = (float*)carve((size_t)NN * HEADS * 4);
    unsigned* mxk = (unsigned*)carve((size_t)NN * HEADS * 4);
    float*    Zb  = (float*)carve((size_t)NN * HEADS * 4);
    float*    sums = (float*)carve((size_t)GG * HID * 4);
    float*    cnt  = (float*)carve((size_t)GG * 4);
    unsigned* mxp  = (unsigned*)carve((size_t)GG * HID * 4);
    float*    pooled = (float*)carve((size_t)GG * 3 * HID * 4);
    (void)ws_size; (void)in_sizes; (void)n_in; (void)out_size;

    // ---- pack weights into WMMA B-fragment layout ----
    k_pack_W<<<gsz(4 * 16 * 256), WG, 0, stream>>>(L[0].W, Wpk1, 128);
    k_pack_W<<<gsz(8 * 16 * 256), WG, 0, stream>>>(L[1].W, Wpk2, 256);
    k_pack_W<<<gsz(8 * 16 * 256), WG, 0, stream>>>(L[2].W, Wpk3, 256);

    const float* feat = x;
    const unsigned* wpk[3] = {Wpk1, Wpk2, Wpk3};
    int dims[3] = {128, 256, 256};

    for (int l = 0; l < 3; ++l) {
        int D = dims[l], KT = D >> 5;
        // GEMM: hP = feat @ W  (WMMA bf16, double-buffered)
        k_pack_A<<<gsz((long long)RT * KT * 256), WG, 0, stream>>>(feat, Apk, NN, D);
        k_gemm<<<dim3((RT + 1) / 2), WG, 0, stream>>>(Apk, wpk[l], hP, NN, KT);
        // attention scores
        k_scores<<<gsz((long long)NN * HEADS), WG, 0, stream>>>(hP, L[l].as_, L[l].ad, ss, sd, NN);
        // segment softmax + aggregation
        k_fill_u32<<<gsz(NN * HEADS), WG, 0, stream>>>(mxk, ENC_NEG_INF, NN * HEADS);
        k_fill_u32<<<gsz(NN * HEADS), WG, 0, stream>>>((unsigned*)Zb, 0u, NN * HEADS);
        k_fill_u32<<<gsz((long long)NN * HID), WG, 0, stream>>>((unsigned*)hO, 0u, NN * HID);
        k_edge_max<<<gsz((long long)(EE + NN) * HEADS), WG, 0, stream>>>(ei, ss, sd, mxk);
        k_edge_exp<<<gsz((long long)(EE + NN) * HEADS), WG, 0, stream>>>(ei, ss, sd, mxk, ebuf, Zb);
        k_edge_agg<<<gsz((long long)(EE + NN) * (HID / 4)), WG, 0, stream>>>(ei, ebuf, Zb, hP, hO);
        // bias + BN + residual + ELU
        k_post<<<gsz((long long)NN * HID), WG, 0, stream>>>(
            hO, L[l].b, L[l].g, L[l].be, L[l].m, L[l].v, feat, hA, NN, l > 0 ? 1 : 0);
        feat = hA;
    }

    // ---- tri-pooling + heads ----
    k_fill_u32<<<gsz(GG * HID), WG, 0, stream>>>((unsigned*)sums, 0u, GG * HID);
    k_fill_u32<<<gsz(GG), WG, 0, stream>>>((unsigned*)cnt, 0u, GG);
    k_fill_u32<<<gsz(GG * HID), WG, 0, stream>>>(mxp, ENC_NEG_INF, GG * HID);
    k_pool_cnt<<<gsz(NN), WG, 0, stream>>>(batch, cnt, NN);
    k_pool_acc<<<gsz((long long)NN * HID), WG, 0, stream>>>(hA, batch, sums, mxp, NN);
    k_pooled<<<gsz(GG * HID), WG, 0, stream>>>(sums, cnt, mxp, pooled);
    k_heads<<<gsz(GG * 6), WG, 0, stream>>>(pooled, ow0, ob0, ow1, ob1, (float*)d_out);
}
